// MultiHeadAttentionQuantum_65481071399214
// MI455X (gfx1250) — matrix operand order repaired and checked by reading, not compiled, not run on hardware
//
#include <hip/hip_runtime.h>
#include <hip/hip_bf16.h>

typedef __attribute__((ext_vector_type(2)))  float    v2f;
typedef __attribute__((ext_vector_type(8)))  float    v8f;
typedef __attribute__((ext_vector_type(16))) _Float16 v16h;
typedef __attribute__((ext_vector_type(8)))  _Float16 h8;
typedef __attribute__((ext_vector_type(2)))  _Float16 h2;

#define N_B 4
#define N_S 1024
#define N_H 32
#define N_D 8
#define N_E 256

// ---------------------------------------------------------------------------
// Kernel 1: closed-form quantum heads.
//   c_j = cos(x_j + theta_j);  z_w = c_0*...*c_w (w>=1),  z_0 = c_1*...*c_7.
// ws_q[head][d][s] fp32 (natural key order).
// ws_v[head][d][s'] fp16 with keys PERMUTED inside each 32-block:
//   actual key a -> slot 2a (a<16) / 2(a-16)+1 (a>=16), matching the packed
//   P-tile store in the attention kernel (K is a contraction dim; any
//   permutation is legal if A and B agree).
// ---------------------------------------------------------------------------
__global__ __launch_bounds__(256) void qheads_kernel(
    const float* __restrict__ x, const float* __restrict__ theta,
    float* __restrict__ ws_q, _Float16* __restrict__ ws_v) {
  const int row = blockIdx.x * 256 + threadIdx.x;      // (b, s, h) row of [B*S*H, 8]
  const float* xr = x + (size_t)row * 8;
  float4 x0 = ((const float4*)xr)[0];
  float4 x1 = ((const float4*)xr)[1];
  float c[8];
  c[0] = __cosf(x0.x + theta[0]);
  c[1] = __cosf(x0.y + theta[1]);
  c[2] = __cosf(x0.z + theta[2]);
  c[3] = __cosf(x0.w + theta[3]);
  c[4] = __cosf(x1.x + theta[4]);
  c[5] = __cosf(x1.y + theta[5]);
  c[6] = __cosf(x1.z + theta[6]);
  c[7] = __cosf(x1.w + theta[7]);
  float z[8];
  z[0] = c[1] * c[2] * c[3] * c[4] * c[5] * c[6] * c[7];
  float p = c[0];
#pragma unroll
  for (int w = 1; w < 8; ++w) { p *= c[w]; z[w] = p; }

  const int head = ((row >> 15) << 5) | (row & 31);    // b*32 + h
  const int s    = (row >> 5) & 1023;
  const int a    = s & 31;
  const int sp   = (s & ~31) | ((a < 16) ? (2 * a) : (2 * (a - 16) + 1));
  const size_t bq = (size_t)head * 8192 + s;           // [head][d][s], stride 1024
  const size_t bv = (size_t)head * 8192 + sp;
#pragma unroll
  for (int d = 0; d < 8; ++d) {
    ws_q[bq + (size_t)d * 1024] = z[d];
    ws_v[bv + (size_t)d * 1024] = (_Float16)z[d];
  }
}

// ---------------------------------------------------------------------------
// Kernel 2: w_combine fp32 -> fp16 (row-major preserved: W[n][k])
// ---------------------------------------------------------------------------
__global__ __launch_bounds__(256) void wconv_kernel(
    const float* __restrict__ w, _Float16* __restrict__ wf) {
  const int i = blockIdx.x * 256 + threadIdx.x;
  wf[i] = (_Float16)w[i];
}

// ---------------------------------------------------------------------------
// Kernel 3: per-head attention. 1 block per (b,h); 8 waves; each wave owns
// 8 query tiles, processed as 2 groups of 4 so the score B-operands and the
// V B-operand are loaded once per K step and shared by 4 query tiles.
//
// |score| <= sqrt(8) so softmax needs no max subtraction; the row normalizer
// comes free from the P@V WMMA via a ones-column in V column 8.
// exp uses v_exp_f32 (=exp2) directly: Q is pre-scaled by log2(e)/sqrt(8).
//
// QK^T: 2x v_wmma_f32_16x16x4_f32 per 16x16 tile (exact fp32, K=8).
// P@V : v_wmma_f32_16x16x32_f16, 32 keys per instruction.
// ---------------------------------------------------------------------------
__global__ __launch_bounds__(256) void qattn_kernel(
    const float* __restrict__ ws_q, const _Float16* __restrict__ ws_v,
    _Float16* __restrict__ attn_out) {
  __shared__ __align__(16) float    qT[8][1024];      // qh^T fp32 (32 KB)
  __shared__ __align__(16) _Float16 vT[16][1032];     // rows 0-7: qh^T fp16 (K-permuted);
                                                      // row 8: ones; 9-15: zeros; padded stride
  __shared__ __align__(16) _Float16 pbuf[8][16][32];  // per-wave P tile (8 KB)

  const int head = blockIdx.x;           // b*32 + h
  const int tid  = threadIdx.x;
  const int wave = tid >> 5;
  const int lane = tid & 31;
  const int n    = lane & 15;            // tile row (A) / tile col (B,C)
  const int hi   = lane >> 4;

  // cooperative LDS fill
  {
    const float4* g4 = (const float4*)(ws_q + (size_t)head * 8192);
    float4* q4 = (float4*)&qT[0][0];
    for (int i = tid; i < 2048; i += 256) q4[i] = g4[i];
    const float4* gv4 = (const float4*)(ws_v + (size_t)head * 8192);
    for (int i = tid; i < 1024; i += 256) {
      const int d = i >> 7, cch = i & 127;             // 128 float4-chunks per row
      ((float4*)&vT[d][0])[cch] = gv4[i];
    }
    for (int i = tid; i < 8192; i += 256) {
      const int d = 8 + (i >> 10), s = i & 1023;
      vT[d][s] = (d == 8) ? (_Float16)1.0f : (_Float16)0.0f;
    }
  }
  __syncthreads();

  const int b = head >> 5, h = head & 31;
  const float slog2e = 0.51012725222f;   // log2(e) / sqrt(8)
  _Float16 (*pb)[32] = pbuf[wave];
  const v8f vzero = {};

  for (int g = 0; g < 2; ++g) {
    int q0[4];
    v2f a1[4], a2[4];
    v8f acc[4];
#pragma unroll
    for (int j = 0; j < 4; ++j) {
      q0[j] = (wave + 8 * (4 * g + j)) * 16;
      a1[j].x = qT[2 * hi + 0][q0[j] + n] * slog2e;
      a1[j].y = qT[2 * hi + 1][q0[j] + n] * slog2e;
      a2[j].x = qT[2 * hi + 4][q0[j] + n] * slog2e;
      a2[j].y = qT[2 * hi + 5][q0[j] + n] * slog2e;
      acc[j] = vzero;                    // cols 0-7: P@V ; col 8: row sum
    }

    for (int k0 = 0; k0 < 1024; k0 += 32) {
      // score B-operands (keys k0..k0+15 and k0+16..k0+31), shared by 4 tiles
      v2f bA, bB, bC, bD;
      bA.x = qT[2 * hi + 0][k0 + n];      bA.y = qT[2 * hi + 1][k0 + n];
      bB.x = qT[2 * hi + 4][k0 + n];      bB.y = qT[2 * hi + 5][k0 + n];
      bC.x = qT[2 * hi + 0][k0 + 16 + n]; bC.y = qT[2 * hi + 1][k0 + 16 + n];
      bD.x = qT[2 * hi + 4][k0 + 16 + n]; bD.y = qT[2 * hi + 5][k0 + 16 + n];
      // V B-operand (K-permuted rows; col 8 = ones, 9-15 = zeros), shared
      h8 vlo = *(const h8*)&vT[n][k0 + 8 * hi];
      h8 vhi = *(const h8*)&vT[n][k0 + 16 + 8 * hi];
      const v16h bp = __builtin_shufflevector(vlo, vhi,
          0, 1, 2, 3, 4, 5, 6, 7, 8, 9, 10, 11, 12, 13, 14, 15);

#pragma unroll
      for (int j = 0; j < 4; ++j) {
        v8f c1 = __builtin_amdgcn_wmma_f32_16x16x4_f32(false, a1[j], false, bA, (short)0, vzero, false, false);
        c1     = __builtin_amdgcn_wmma_f32_16x16x4_f32(false, a2[j], false, bB, (short)0, c1,    false, false);
        v8f c2 = __builtin_amdgcn_wmma_f32_16x16x4_f32(false, a1[j], false, bC, (short)0, vzero, false, false);
        c2     = __builtin_amdgcn_wmma_f32_16x16x4_f32(false, a2[j], false, bD, (short)0, c2,    false, false);

        // p = 2^(s*log2e); packed b32 store into K-permuted P tile
#pragma unroll
        for (int r = 0; r < 8; ++r) {
          h2 pp;
          pp.x = (_Float16)__builtin_amdgcn_exp2f(c1[r]);
          pp.y = (_Float16)__builtin_amdgcn_exp2f(c2[r]);
          *(h2*)&pb[r + 8 * hi][2 * n] = pp;
        }
        asm volatile("" ::: "memory");   // same-wave LDS is in-order; block reordering

        // P as A operand (16x32 f16 layout), two b128 LDS loads
        h8 plo = *(const h8*)&pb[n][8 * hi];
        h8 phi = *(const h8*)&pb[n][16 + 8 * hi];
        const v16h ap = __builtin_shufflevector(plo, phi,
            0, 1, 2, 3, 4, 5, 6, 7, 8, 9, 10, 11, 12, 13, 14, 15);

        acc[j] = __builtin_amdgcn_wmma_f32_16x16x32_f16(false, ap, false, bp, (short)0, acc[j], false, false);
      }
    }

    // normalize by row sum (column 8 = lane 16*hi+8) and store fp16
#pragma unroll
    for (int j = 0; j < 4; ++j) {
#pragma unroll
      for (int r = 0; r < 8; ++r) {
        const float l    = __shfl(acc[j][r], 16 * hi + 8, 32);
        const float rinv = __builtin_amdgcn_rcpf(l);
        if (n < 8) {
          attn_out[((size_t)(b * N_S + q0[j] + r + 8 * hi) * N_H + h) * 8 + n] =
              (_Float16)(acc[j][r] * rinv);
        }
      }
    }
  }
}

// ---------------------------------------------------------------------------
// Kernel 4: combine GEMM  out[4096,256] = A[4096,256] @ W^T, f16 WMMA, f32 acc.
// B-operand B[k][n] = W[n][k] -> contiguous K runs in row-major W.
// One 16x16 output tile per wave; 8 k-steps of 32; b128 global loads.
// ---------------------------------------------------------------------------
__global__ __launch_bounds__(256) void combine_kernel(
    const _Float16* __restrict__ A, const _Float16* __restrict__ Wf,
    float* __restrict__ out) {
  const int tid  = threadIdx.x;
  const int wave = tid >> 5;
  const int lane = tid & 31;
  const int n    = lane & 15;
  const int hi   = lane >> 4;
  const int tile = blockIdx.x * 8 + wave;      // 4096 tiles: 256 (M) x 16 (N)
  const int m0   = (tile >> 4) * 16;
  const int n0   = (tile & 15) * 16;
  const size_t abase = (size_t)(m0 + n) * N_E;
  const size_t wbase = (size_t)(n0 + n) * N_E;

  v8f acc = {};
  for (int k0 = 0; k0 < N_E; k0 += 32) {
    h8 alo = *(const h8*)(A + abase + k0 + 8 * hi);
    h8 ahi = *(const h8*)(A + abase + k0 + 16 + 8 * hi);
    const v16h ap = __builtin_shufflevector(alo, ahi,
        0, 1, 2, 3, 4, 5, 6, 7, 8, 9, 10, 11, 12, 13, 14, 15);
    h8 blo = *(const h8*)(Wf + wbase + k0 + 8 * hi);
    h8 bhi = *(const h8*)(Wf + wbase + k0 + 16 + 8 * hi);
    const v16h bp = __builtin_shufflevector(blo, bhi,
        0, 1, 2, 3, 4, 5, 6, 7, 8, 9, 10, 11, 12, 13, 14, 15);
    acc = __builtin_amdgcn_wmma_f32_16x16x32_f16(false, ap, false, bp, (short)0, acc, false, false);
  }
#pragma unroll
  for (int r = 0; r < 8; ++r)
    out[(size_t)(m0 + r + 8 * hi) * N_E + n0 + n] = acc[r];
}

// ---------------------------------------------------------------------------
// Workspace layout (bytes):
//   [0,        4 MiB)  ws_q   : fp32 qh, [head][d][s]
//   [4 MiB,    6 MiB)  ws_v   : fp16 qh, [head][d][s'] (K-permuted per 32)
//   [6 MiB,    8 MiB)  attn   : fp16 attention output, [4096][256]
//   [8 MiB, 8.125 MiB) wf16   : fp16 w_combine, row-major [256][256]
// ---------------------------------------------------------------------------
extern "C" void kernel_launch(void* const* d_in, const int* in_sizes, int n_in,
                              void* d_out, int out_size, void* d_ws, size_t ws_size,
                              hipStream_t stream) {
  const float* x     = (const float*)d_in[0];
  const float* theta = (const float*)d_in[1];
  const float* w     = (const float*)d_in[2];
  float* out         = (float*)d_out;

  char* ws = (char*)d_ws;
  float*    ws_q = (float*)(ws);
  _Float16* ws_v = (_Float16*)(ws + (size_t)4 * 1024 * 1024);
  _Float16* attn = (_Float16*)(ws + (size_t)6 * 1024 * 1024);
  _Float16* wf16 = (_Float16*)(ws + (size_t)8 * 1024 * 1024);

  qheads_kernel <<<512, 256, 0, stream>>>(x, theta, ws_q, ws_v);
  wconv_kernel  <<<256, 256, 0, stream>>>(w, wf16);
  qattn_kernel  <<<128, 256, 0, stream>>>(ws_q, ws_v, attn);
  combine_kernel<<<512, 256, 0, stream>>>(attn, wf16, out);
  (void)in_sizes; (void)n_in; (void)out_size; (void)ws_size;
}
